// VariableAnnuity_10505490006644
// MI455X (gfx1250) — matrix operand order repaired and checked by reading, not compiled, not run on hardware
//
#include <hip/hip_runtime.h>
#include <cstdint>

typedef __attribute__((ext_vector_type(16))) _Float16 v16h;
typedef __attribute__((ext_vector_type(8)))  float    v8f;

#define BATCH     65536
#define NSTEPS    60
#define HIDDEN    128
#define DT_C      (1.0f/12.0f)
#define LAM_C     0.01f
#define TEXP_C    5.0f
#define PRIN_C    100.0f
#define GMDB_C    100.0f
#define FEE_C     0.0196f

#define WAVES_PER_BLOCK 4
#define PATHS_PER_WAVE  32
#define THREADS (WAVES_PER_BLOCK * 32)
#define PATHS_PER_BLOCK (WAVES_PER_BLOCK * PATHS_PER_WAVE)

union FragAB { v16h v; uint32_t u[8]; };

__launch_bounds__(THREADS, 1)
__global__ void va_pnl_kernel(const float* __restrict__ spots,
                              const float* __restrict__ W1,
                              const float* __restrict__ b1,
                              const float* __restrict__ W2,
                              const float* __restrict__ b2,
                              const float* __restrict__ W3,
                              const float* __restrict__ b3,
                              float* __restrict__ out)
{
    // 32 KB: W2 transposed to [n][k] in f16 so B-fragments are K-contiguous.
    __shared__ _Float16 s_W2t[HIDDEN * HIDDEN];
    // 32 KB: per-wave layer-1 activations, row-major [path][k] f16.
    // (front of each wave's slab is reused as fp32 layer-3 staging once h1 is dead)
    __shared__ _Float16 s_h1[WAVES_PER_BLOCK][PATHS_PER_WAVE * HIDDEN];

    const int tid  = threadIdx.x;
    const int wave = tid >> 5;
    const int lane = tid & 31;
    const int half = lane >> 4;   // which half-wave (WMMA layout)
    const int l15  = lane & 15;

    // ---- one-time: transpose W2 (f32 row-major [k][n]) -> f16 [n][k] ----
    for (int idx = tid; idx < HIDDEN * HIDDEN; idx += THREADS) {
        const int n = idx >> 7;
        const int k = idx & (HIDDEN - 1);
        s_W2t[idx] = (_Float16)W2[k * HIDDEN + n];
    }
    __syncthreads();

    // ---- one-time: layer-1 weights, register resident (lane owns j pairs) --
    const int jA = 2 * lane;          // covers j = 0..63
    const int jB = 64 + 2 * lane;     // covers j = 64..127
    const float w0A0 = W1[jA],          w0A1 = W1[jA + 1];
    const float w1A0 = W1[HIDDEN + jA], w1A1 = W1[HIDDEN + jA + 1];
    const float bA0  = b1[jA],          bA1  = b1[jA + 1];
    const float w0B0 = W1[jB],          w0B1 = W1[jB + 1];
    const float w1B0 = W1[HIDDEN + jB], w1B1 = W1[HIDDEN + jB + 1];
    const float bB0  = b1[jB],          bB1  = b1[jB + 1];

    // ---- one-time: b2 / W3 slices matching WMMA C-layout column N = l15 ----
    float b2reg[8], w3reg[8];
#pragma unroll
    for (int nt = 0; nt < 8; ++nt) {
        b2reg[nt] = b2[nt * 16 + l15];
        w3reg[nt] = W3[nt * 16 + l15];
    }
    const float b3v = b3[0];

    const int gpath = blockIdx.x * PATHS_PER_BLOCK + wave * PATHS_PER_WAVE + lane;

    float pnl  = 0.0f;
    float spot = spots[gpath];  // t = 0 row

    for (int step = 0; step < NSTEPS; ++step) {
        const float t = (float)step * DT_C;
        const float spot_next = spots[(step + 1) * BATCH + gpath];

        // ================= layer 1 (K=2, VALU) =================
        // h1[p][j] = relu(spot_p * W1[0][j] + t * W1[1][j] + b1[j])
        const float cA0 = fmaf(t, w1A0, bA0), cA1 = fmaf(t, w1A1, bA1);
        const float cB0 = fmaf(t, w1B0, bB0), cB1 = fmaf(t, w1B1, bB1);
        const uint32_t spot_bits = __float_as_uint(spot);
#pragma unroll
        for (int p = 0; p < PATHS_PER_WAVE; ++p) {
            const float sp = __uint_as_float(__builtin_amdgcn_readlane(spot_bits, p));
            const float hA0 = fmaxf(fmaf(sp, w0A0, cA0), 0.0f);
            const float hA1 = fmaxf(fmaf(sp, w0A1, cA1), 0.0f);
            const float hB0 = fmaxf(fmaf(sp, w0B0, cB0), 0.0f);
            const float hB1 = fmaxf(fmaf(sp, w0B1, cB1), 0.0f);
            const uint32_t pA =
                __builtin_bit_cast(uint32_t, __builtin_amdgcn_cvt_pkrtz(hA0, hA1));
            const uint32_t pB =
                __builtin_bit_cast(uint32_t, __builtin_amdgcn_cvt_pkrtz(hB0, hB1));
            *reinterpret_cast<uint32_t*>(&s_h1[wave][p * HIDDEN + jA]) = pA;
            *reinterpret_cast<uint32_t*>(&s_h1[wave][p * HIDDEN + jB]) = pB;
        }
        __syncthreads();

        // ================= layer 2 (WMMA f16 -> f32) =================
        // A fragments: 2 M-tiles x 4 K-tiles, 16x32 f16 each, per ISA layout:
        // lane = half*16 + (M - m0); word w holds K = (w>>2)*16 + half*8 + (w&3)*2
        FragAB afrag[2][4];
#pragma unroll
        for (int mt = 0; mt < 2; ++mt)
#pragma unroll
            for (int kt = 0; kt < 4; ++kt)
#pragma unroll
                for (int w = 0; w < 8; ++w) {
                    const int row = mt * 16 + l15;
                    const int K   = kt * 32 + ((w >> 2) * 16) + half * 8 + (w & 3) * 2;
                    afrag[mt][kt].u[w] = *reinterpret_cast<const uint32_t*>(
                        &s_h1[wave][row * HIDDEN + K]);
                }

        v8f acc[2][8];
#pragma unroll
        for (int mt = 0; mt < 2; ++mt)
#pragma unroll
            for (int nt = 0; nt < 8; ++nt)
                acc[mt][nt] = (v8f){};

#pragma unroll
        for (int nt = 0; nt < 8; ++nt) {
#pragma unroll
            for (int kt = 0; kt < 4; ++kt) {
                // B fragment 32x16 f16: lane = half*16 + N; word w: K = (w>>2)*16 + half*8 + (w&3)*2
                FragAB bfrag;
#pragma unroll
                for (int w = 0; w < 8; ++w) {
                    const int N = nt * 16 + l15;
                    const int K = kt * 32 + ((w >> 2) * 16) + half * 8 + (w & 3) * 2;
                    bfrag.u[w] = *reinterpret_cast<const uint32_t*>(
                        &s_W2t[N * HIDDEN + K]);
                }
#pragma unroll
                for (int mt = 0; mt < 2; ++mt) {
                    acc[mt][nt] = __builtin_amdgcn_wmma_f32_16x16x32_f16(
                        false, afrag[mt][kt].v,
                        false, bfrag.v,
                        (short)0, acc[mt][nt],
                        false, false);
                }
            }
        }

        // ============ layer 3 (dot with W3) + bias/relu fused ============
        // C layout: lane holds column N = l15, rows M = r + half*8 (r = VGPR idx)
        float* s_outw = reinterpret_cast<float*>(&s_h1[wave][0]);  // h1 is dead now
#pragma unroll
        for (int mt = 0; mt < 2; ++mt) {
            float s[8];
#pragma unroll
            for (int r = 0; r < 8; ++r) s[r] = 0.0f;
#pragma unroll
            for (int nt = 0; nt < 8; ++nt)
#pragma unroll
                for (int r = 0; r < 8; ++r) {
                    const float h2 = fmaxf(acc[mt][nt][r] + b2reg[nt], 0.0f);
                    s[r] = fmaf(h2, w3reg[nt], s[r]);
                }
            // reduce across the 16 columns (xor masks stay within each half)
#pragma unroll
            for (int mask = 1; mask <= 8; mask <<= 1)
#pragma unroll
                for (int r = 0; r < 8; ++r)
                    s[r] += __shfl_xor(s[r], mask, 32);
            if (l15 == 0) {
#pragma unroll
                for (int r = 0; r < 8; ++r)
                    s_outw[mt * 16 + half * 8 + r] = s[r];
            }
        }
        __syncthreads();

        // ================= financial PnL update (per path = per lane) ======
        const float o     = s_outw[lane] + b3v;
        float delta = -(o * o);
        delta = delta * fminf(__expf(-0.01f * delta), 1.0f);
        delta = delta * ((1.0f - __expf(-LAM_C * (TEXP_C - t))) * PRIN_C);
        const float account = PRIN_C * spot * __expf(-FEE_C * t);
        const float el      = __expf(-LAM_C * t);
        const float fee     = FEE_C * DT_C * account * el;
        const float payout  = LAM_C * DT_C * fmaxf(GMDB_C - account, 0.0f) * el;
        pnl += fee - payout + delta * (spot_next - spot);
        spot = spot_next;
        __syncthreads();  // s_h1 / s_outw recycled next step
    }

    out[gpath] = pnl;
}

extern "C" void kernel_launch(void* const* d_in, const int* in_sizes, int n_in,
                              void* d_out, int out_size, void* d_ws, size_t ws_size,
                              hipStream_t stream) {
    (void)in_sizes; (void)n_in; (void)d_ws; (void)ws_size; (void)out_size;
    const float* spots = (const float*)d_in[0];
    const float* W1    = (const float*)d_in[1];
    const float* b1    = (const float*)d_in[2];
    const float* W2    = (const float*)d_in[3];
    const float* b2    = (const float*)d_in[4];
    const float* W3    = (const float*)d_in[5];
    const float* b3    = (const float*)d_in[6];
    float* out = (float*)d_out;

    const int blocks = BATCH / PATHS_PER_BLOCK;  // 512
    va_pnl_kernel<<<blocks, THREADS, 0, stream>>>(spots, W1, b1, W2, b2, W3, b3, out);
}
